// VMambaSS2D_26869315404386
// MI455X (gfx1250) — matrix operand order repaired and checked by reading, not compiled, not run on hardware
//
#include <hip/hip_runtime.h>
#include <hip/hip_bf16.h>
#include <math.h>

// ---------------------------------------------------------------------------
// VMamba SS2D for MI455X (gfx1250): WMMA f16->f32 GEMMs + 3-phase chunked scan
// with async (memory -> LDS) double-buffered operand staging in the scan.
// ---------------------------------------------------------------------------
// Shapes: B=2, H=W=64, L=4096, D_MODEL=96, D_INNER=192, N=16, R=6, K=4
// ---------------------------------------------------------------------------

typedef __attribute__((ext_vector_type(16))) _Float16 v16h;
typedef __attribute__((ext_vector_type(8)))  float    v8f;

#define BATCH   2
#define HH      64
#define WW      64
#define LL      4096
#define DMODEL  96
#define DINNER  192
#define NSTATE  16
#define RRANK   6
#define KDIR    4
#define GW      40          // padded 38 -> 40 row width for G buffer
#define NCHUNK  16
#define CHUNK   256
#define STAGE   136         // 128 B/C floats + 6 dt floats (padded to 136)

// Workspace layout (float offsets)
#define XZ_OFF     ((size_t)0)                      // 8192*384
#define XC_OFF     (XZ_OFF + (size_t)8192*384)      // 8192*192
#define G_OFF      (XC_OFF + (size_t)8192*192)      // 8*4096*40
#define PS_OFF     (G_OFF  + (size_t)8*4096*40)     // 8*16*192*32
#define CARRY_OFF  (PS_OFF + (size_t)8*16*192*32)   // 8*16*192*16
#define OUTY_OFF   (CARRY_OFF + (size_t)8*16*192*16)// 8*4096*192
#define YG_OFF     (OUTY_OFF + (size_t)8*4096*192)  // 8192*192

// ---------------------------------------------------------------------------
// direction permutation: canonical index of step l in direction k
// k=0: l ; k=1: transpose(h,w) ; k=2: reversed ; k=3: reversed+transposed
// ---------------------------------------------------------------------------
__device__ __forceinline__ int perm_idx(int k, int l) {
    int t = (k >= 2) ? (LL - 1 - l) : l;
    if (k & 1) t = ((t & 63) << 6) | (t >> 6);
    return t;
}

// ---------------------------------------------------------------------------
// Generic WMMA GEMM: out[m*ldo+n] = sum_k A[m*lda+k] * W[n*ldw+k]   (X @ W^T)
// One 16x16 output tile per wave; K must be a multiple of 32.
// A fragment k order used here (16-bit A 16x32): lane m=lane%16, K = (lane/16)*16 + i
// B fragment k order used here (same as A): lane n=lane%16, K = (lane/16)*16 + i
// C/D layout: element (M = r + 8*(lane/16), N = lane%16) in vgpr r
// ---------------------------------------------------------------------------
__global__ void gemm_wmma_xt(const float* __restrict__ A, const float* __restrict__ W,
                             float* __restrict__ out,
                             int M, int N, int Kdim, int lda, int ldw, int ldo) {
    int gw   = (blockIdx.x * blockDim.x + threadIdx.x) >> 5;
    int lane = threadIdx.x & 31;
    int tiles_n = (N + 15) >> 4;
    int tiles_m = (M + 15) >> 4;
    if (gw >= tiles_m * tiles_n) return;           // wave-uniform
    int m0 = (gw / tiles_n) << 4;
    int n0 = (gw % tiles_n) << 4;
    int m  = m0 + (lane & 15);
    int n  = n0 + (lane & 15);
    int koffA = (lane >> 4) * 16;
    int koffB = (lane >> 4) * 16;
    v8f acc = {0.f, 0.f, 0.f, 0.f, 0.f, 0.f, 0.f, 0.f};
    for (int k0 = 0; k0 < Kdim; k0 += 32) {
        v16h a, b;
        if (m < M) {
            const float* pa = A + (size_t)m * lda + k0 + koffA;
            #pragma unroll
            for (int i = 0; i < 8; ++i) a[i]     = (_Float16)pa[i];
            #pragma unroll
            for (int i = 0; i < 8; ++i) a[8 + i] = (_Float16)pa[8 + i];
        } else {
            #pragma unroll
            for (int i = 0; i < 16; ++i) a[i] = (_Float16)0.f;
        }
        if (n < N) {
            const float* pb = W + (size_t)n * ldw + k0 + koffB;
            #pragma unroll
            for (int i = 0; i < 16; ++i) b[i] = (_Float16)pb[i];
        } else {
            #pragma unroll
            for (int i = 0; i < 16; ++i) b[i] = (_Float16)0.f;
        }
        acc = __builtin_amdgcn_wmma_f32_16x16x32_f16(false, a, false, b,
                                                     (short)0, acc, false, false);
    }
    if (n < N) {
        int rbase = m0 + ((lane >> 4) << 3);
        #pragma unroll
        for (int r = 0; r < 8; ++r) {
            int mm = rbase + r;
            if (mm < M) out[(size_t)mm * ldo + n] = acc[r];
        }
    }
}

// ---------------------------------------------------------------------------
// Depthwise 3x3 conv + bias + SiLU.  Reads xx slice of xz (B,L,384), writes
// xc (B,L,192) in canonical row-major (d fastest) layout.
// ---------------------------------------------------------------------------
__global__ void conv_silu(const float* __restrict__ xz, const float* __restrict__ cw,
                          const float* __restrict__ cb, float* __restrict__ xcout) {
    int idx = blockIdx.x * blockDim.x + threadIdx.x;
    if (idx >= BATCH * LL * DINNER) return;
    int d = idx % DINNER;
    int l = (idx / DINNER) % LL;
    int b = idx / (DINNER * LL);
    int h = l >> 6, w = l & 63;
    float acc = cb[d];
    #pragma unroll
    for (int dh = -1; dh <= 1; ++dh) {
        int hh = h + dh;
        if (hh < 0 || hh > 63) continue;
        #pragma unroll
        for (int dw = -1; dw <= 1; ++dw) {
            int ww2 = w + dw;
            if (ww2 < 0 || ww2 > 63) continue;
            acc += xz[((size_t)(b * LL + (hh << 6) + ww2)) * 384 + d] *
                   cw[d * 9 + (dh + 1) * 3 + (dw + 1)];
        }
    }
    xcout[(size_t)idx] = acc / (1.f + __expf(-acc));
}

// ---------------------------------------------------------------------------
// Per-direction projection GEMM:  G_k[b, m, c] = sum_d xpw[k,c,d] * xc[b,m,d]
// (direction k's x_dbl at step l is G_k[b, perm_k(l), :]).
// Output layout: G[((b*4+k)*4096 + m)*40 + c], c in [0,38), padded to 40.
// One 16(c) x 16(m) tile per wave.
// ---------------------------------------------------------------------------
__global__ void xdbl_gemm_wmma(const float* __restrict__ xc, const float* __restrict__ xpw,
                               float* __restrict__ G) {
    int gw   = (blockIdx.x * blockDim.x + threadIdx.x) >> 5;
    int lane = threadIdx.x & 31;
    int ctile   = gw % 3;
    int tmp     = gw / 3;
    int coltile = tmp % 256;
    int bk      = tmp / 256;
    if (bk >= 8) return;
    int b = bk >> 2, k = bk & 3;
    int c0 = ctile << 4, col0 = coltile << 4;
    const float* A   = xpw + (size_t)k * 38 * DINNER;          // (38 x 192)
    const float* xcb = xc  + (size_t)b * LL * DINNER;
    int c   = c0 + (lane & 15);
    int col = col0 + (lane & 15);
    int koffA = (lane >> 4) * 16;
    int koffB = (lane >> 4) * 16;
    v8f acc = {0.f, 0.f, 0.f, 0.f, 0.f, 0.f, 0.f, 0.f};
    for (int k0 = 0; k0 < DINNER; k0 += 32) {
        v16h a, bfr;
        if (c < 38) {
            const float* pa = A + (size_t)c * DINNER + k0 + koffA;
            #pragma unroll
            for (int i = 0; i < 8; ++i) a[i]     = (_Float16)pa[i];
            #pragma unroll
            for (int i = 0; i < 8; ++i) a[8 + i] = (_Float16)pa[8 + i];
        } else {
            #pragma unroll
            for (int i = 0; i < 16; ++i) a[i] = (_Float16)0.f;
        }
        {
            const float* pb = xcb + (size_t)col * DINNER + k0 + koffB;
            #pragma unroll
            for (int i = 0; i < 16; ++i) bfr[i] = (_Float16)pb[i];
        }
        acc = __builtin_amdgcn_wmma_f32_16x16x32_f16(false, a, false, bfr,
                                                     (short)0, acc, false, false);
    }
    // transposed store into (m, c) layout
    int cb2 = c0 + ((lane >> 4) << 3);
    float* gb = G + ((size_t)bk * LL + col) * GW;
    #pragma unroll
    for (int r = 0; r < 8; ++r) {
        int cc = cb2 + r;
        if (cc < GW) gb[cc] = acc[r];
    }
}

// ---------------------------------------------------------------------------
// Async staging of one scan step's operands into LDS via CDNA5
// GLOBAL_LOAD_ASYNC_TO_LDS_B32 (tracked with ASYNCcnt).  Threads 0..127 fetch
// the B/C rows of all 4 directions at their per-direction canonical
// positions; threads 128..133 fetch direction k's 6 dt-rank values.
// The LDS destination address is the low 32 bits of the flat shared pointer
// (LDS aperture keeps the byte offset in addr[31:0]).
// ---------------------------------------------------------------------------
__device__ __forceinline__ void issue_stage_async(const float* __restrict__ Gb, int k, int l,
                                                  float* buf) {
    int tid = threadIdx.x;
    if (tid < 128 + RRANK) {
        const float* src;
        float* dst;
        if (tid < 128) {
            int jj = tid >> 5, c = tid & 31;
            src = Gb + ((size_t)jj * LL + perm_idx(jj, l)) * GW + RRANK + c;
            dst = buf + tid;
        } else {
            int r = tid - 128;
            src = Gb + ((size_t)k * LL + perm_idx(k, l)) * GW + r;
            dst = buf + 128 + r;
        }
        unsigned ldsoff = (unsigned)(unsigned long long)dst;
        asm volatile("global_load_async_to_lds_b32 %0, %1, off"
                     :: "v"(ldsoff), "v"(src)
                     : "memory");
    }
}

__device__ __forceinline__ void wait_async_le1(void) {
    asm volatile("s_wait_asynccnt 0x1" ::: "memory");
}
__device__ __forceinline__ void wait_async_0(void) {
    asm volatile("s_wait_asynccnt 0x0" ::: "memory");
}

// ---------------------------------------------------------------------------
// Scan phase 1: per chunk, compute Pi(a) and final state from zero init.
// grid = 8(bk)*16(chunk), block = 192 (one thread per channel d)
// ---------------------------------------------------------------------------
__global__ void scan_phase1(const float* __restrict__ xc, const float* __restrict__ G,
                            const float* __restrict__ dtw, const float* __restrict__ dtb,
                            const float* __restrict__ A_logs, float* __restrict__ PS) {
    int bk = blockIdx.x >> 4;
    int chunk = blockIdx.x & 15;
    int b = bk >> 2, k = bk & 3;
    int d = threadIdx.x;
    float wreg[RRANK];
    #pragma unroll
    for (int r = 0; r < RRANK; ++r) wreg[r] = dtw[((size_t)(k * DINNER + d)) * RRANK + r];
    float bias = dtb[k * DINNER + d];
    float An[NSTATE];
    #pragma unroll
    for (int n = 0; n < NSTATE; ++n) An[n] = -__expf(A_logs[(size_t)(k * DINNER + d) * NSTATE + n]);
    float P[NSTATE], S[NSTATE];
    #pragma unroll
    for (int n = 0; n < NSTATE; ++n) { P[n] = 1.f; S[n] = 0.f; }
    int j = d & 3;
    __shared__ float sh[2][STAGE];
    const float* Gb = G + (size_t)(b * KDIR) * LL * GW;
    int lbeg = chunk << 8, lend = lbeg + CHUNK;

    issue_stage_async(Gb, k, lbeg, sh[0]);
    for (int l = lbeg; l < lend; ++l) {
        const float* shc = sh[l & 1];
        if (l + 1 < lend) {
            issue_stage_async(Gb, k, l + 1, sh[(l + 1) & 1]);
            __builtin_prefetch(xc + ((size_t)b * LL + perm_idx(k, l + 1)) * DINNER + d, 0, 0);
            wait_async_le1();       // in-order completion => step-l loads done
        } else {
            wait_async_0();
        }
        __syncthreads();            // step-l staging visible to all waves
        int mk = perm_idx(k, l);
        float u = xc[((size_t)b * LL + mk) * DINNER + d];
        float sr = bias;
        #pragma unroll
        for (int r = 0; r < RRANK; ++r) sr += wreg[r] * shc[128 + r];
        float dt = (sr > 20.f) ? sr : log1pf(__expf(sr));
        float dtu = dt * u;
        const float* Bsh = shc + j * 32;
        #pragma unroll
        for (int n = 0; n < NSTATE; ++n) {
            float a = __expf(dt * An[n]);
            S[n] = a * S[n] + dtu * Bsh[n];
            P[n] *= a;
        }
        __syncthreads();            // reclaim buffer before next async write
    }
    float* o = PS + (((size_t)bk * NCHUNK + chunk) * DINNER + d) * 32;
    #pragma unroll
    for (int n = 0; n < NSTATE; ++n) { o[n] = P[n]; o[16 + n] = S[n]; }
}

// ---------------------------------------------------------------------------
// Scan phase 2: sequential combine across 16 chunks (tiny).
// grid = 8(bk), block = 256; each thread walks a set of (d,n) pairs.
// ---------------------------------------------------------------------------
__global__ void scan_phase2(const float* __restrict__ PS, float* __restrict__ carry) {
    int bk = blockIdx.x;
    for (int idx = threadIdx.x; idx < DINNER * NSTATE; idx += blockDim.x) {
        int d = idx >> 4, n = idx & 15;
        float c = 0.f;
        for (int ch = 0; ch < NCHUNK; ++ch) {
            carry[(((size_t)bk * NCHUNK + ch) * DINNER + d) * NSTATE + n] = c;
            const float* p = PS + (((size_t)bk * NCHUNK + ch) * DINNER + d) * 32;
            c = p[16 + n] + p[n] * c;
        }
    }
}

// ---------------------------------------------------------------------------
// Scan phase 3: replay chunks with correct carry-in, emit y at canonical pos.
// ---------------------------------------------------------------------------
__global__ void scan_phase3(const float* __restrict__ xc, const float* __restrict__ G,
                            const float* __restrict__ dtw, const float* __restrict__ dtb,
                            const float* __restrict__ A_logs, const float* __restrict__ Ds,
                            const float* __restrict__ carry, float* __restrict__ out_y) {
    int bk = blockIdx.x >> 4;
    int chunk = blockIdx.x & 15;
    int b = bk >> 2, k = bk & 3;
    int d = threadIdx.x;
    float wreg[RRANK];
    #pragma unroll
    for (int r = 0; r < RRANK; ++r) wreg[r] = dtw[((size_t)(k * DINNER + d)) * RRANK + r];
    float bias = dtb[k * DINNER + d];
    float Dval = Ds[k * DINNER + d];
    float An[NSTATE], S[NSTATE];
    #pragma unroll
    for (int n = 0; n < NSTATE; ++n) {
        An[n] = -__expf(A_logs[(size_t)(k * DINNER + d) * NSTATE + n]);
        S[n]  = carry[(((size_t)bk * NCHUNK + chunk) * DINNER + d) * NSTATE + n];
    }
    int j = d & 3;
    __shared__ float sh[2][STAGE];
    const float* Gb = G + (size_t)(b * KDIR) * LL * GW;
    int lbeg = chunk << 8, lend = lbeg + CHUNK;

    issue_stage_async(Gb, k, lbeg, sh[0]);
    for (int l = lbeg; l < lend; ++l) {
        const float* shc = sh[l & 1];
        if (l + 1 < lend) {
            issue_stage_async(Gb, k, l + 1, sh[(l + 1) & 1]);
            __builtin_prefetch(xc + ((size_t)b * LL + perm_idx(k, l + 1)) * DINNER + d, 0, 0);
            wait_async_le1();
        } else {
            wait_async_0();
        }
        __syncthreads();
        int mk = perm_idx(k, l);
        float u = xc[((size_t)b * LL + mk) * DINNER + d];
        float sr = bias;
        #pragma unroll
        for (int r = 0; r < RRANK; ++r) sr += wreg[r] * shc[128 + r];
        float dt = (sr > 20.f) ? sr : log1pf(__expf(sr));
        float dtu = dt * u;
        const float* Bsh = shc + j * 32;
        const float* Csh = shc + j * 32 + NSTATE;
        float y = 0.f;
        #pragma unroll
        for (int n = 0; n < NSTATE; ++n) {
            float a = __expf(dt * An[n]);
            S[n] = a * S[n] + dtu * Bsh[n];
            y += S[n] * Csh[n];
        }
        y += u * Dval;
        out_y[((size_t)bk * LL + mk) * DINNER + d] = y;   // canonical position
        __syncthreads();
    }
}

// ---------------------------------------------------------------------------
// Merge 4 directions + LayerNorm + SiLU(z) gating -> yg (B,L,192)
// grid = 8192 (b*L), block = 192
// ---------------------------------------------------------------------------
__global__ void merge_ln_gate(const float* __restrict__ out_y, const float* __restrict__ xz,
                              const float* __restrict__ gamma, const float* __restrict__ beta,
                              float* __restrict__ yg) {
    int bl = blockIdx.x;
    int b = bl >> 12, l = bl & (LL - 1);
    int d = threadIdx.x;
    float y = 0.f;
    #pragma unroll
    for (int k = 0; k < KDIR; ++k)
        y += out_y[(((size_t)(b * KDIR + k) * LL) + l) * DINNER + d];
    float s1 = y, s2 = y * y;
    #pragma unroll
    for (int o = 16; o > 0; o >>= 1) {
        s1 += __shfl_xor(s1, o, 32);
        s2 += __shfl_xor(s2, o, 32);
    }
    __shared__ float w1[6], w2[6];
    int wid = d >> 5, lane = d & 31;
    if (lane == 0) { w1[wid] = s1; w2[wid] = s2; }
    __syncthreads();
    float tot = 0.f, tot2 = 0.f;
    #pragma unroll
    for (int i = 0; i < 6; ++i) { tot += w1[i]; tot2 += w2[i]; }
    float mean = tot * (1.f / DINNER);
    float var  = tot2 * (1.f / DINNER) - mean * mean;
    float yn = (y - mean) * rsqrtf(var + 1e-5f) * gamma[d] + beta[d];
    float z  = xz[(size_t)bl * 384 + DINNER + d];
    yg[(size_t)bl * DINNER + d] = yn * (z / (1.f + __expf(-z)));
}

// ---------------------------------------------------------------------------
extern "C" void kernel_launch(void* const* d_in, const int* in_sizes, int n_in,
                              void* d_out, int out_size, void* d_ws, size_t ws_size,
                              hipStream_t stream) {
    const float* x          = (const float*)d_in[0];
    const float* in_proj_w  = (const float*)d_in[1];
    const float* conv_w     = (const float*)d_in[2];
    const float* conv_b     = (const float*)d_in[3];
    const float* xpw        = (const float*)d_in[4];
    const float* dtw        = (const float*)d_in[5];
    const float* dtb        = (const float*)d_in[6];
    const float* A_logs     = (const float*)d_in[7];
    const float* Ds         = (const float*)d_in[8];
    const float* gamma      = (const float*)d_in[9];
    const float* beta       = (const float*)d_in[10];
    const float* out_proj_w = (const float*)d_in[11];
    float* out = (float*)d_out;
    float* ws  = (float*)d_ws;

    float* xz     = ws + XZ_OFF;
    float* xc     = ws + XC_OFF;
    float* G      = ws + G_OFF;
    float* PS     = ws + PS_OFF;
    float* carryb = ws + CARRY_OFF;
    float* out_y  = ws + OUTY_OFF;
    float* yg     = ws + YG_OFF;

    const int M = BATCH * LL;   // 8192

    // 1. in_proj: xz = x @ in_proj_w^T   (8192 x 384, K=96)
    {
        int waves = (M / 16) * (384 / 16);     // 12288
        gemm_wmma_xt<<<waves / 8, 256, 0, stream>>>(x, in_proj_w, xz,
                                                    M, 384, DMODEL, DMODEL, DMODEL, 384);
    }
    // 2. depthwise conv + SiLU -> xc (B,L,192)
    {
        int total = BATCH * LL * DINNER;
        conv_silu<<<(total + 255) / 256, 256, 0, stream>>>(xz, conv_w, conv_b, xc);
    }
    // 3. per-direction projections G_k = xpw[k] @ xc^T   (38 x 4096 per b,k)
    {
        int waves = 8 * 256 * 3;               // 6144
        xdbl_gemm_wmma<<<waves / 8, 256, 0, stream>>>(xc, xpw, G);
    }
    // 4. chunked selective scan (3-phase linear-recurrence scan)
    scan_phase1<<<8 * NCHUNK, DINNER, 0, stream>>>(xc, G, dtw, dtb, A_logs, PS);
    scan_phase2<<<8, 256, 0, stream>>>(PS, carryb);
    scan_phase3<<<8 * NCHUNK, DINNER, 0, stream>>>(xc, G, dtw, dtb, A_logs, Ds, carryb, out_y);
    // 5. merge directions + LayerNorm + gating
    merge_ln_gate<<<M, DINNER, 0, stream>>>(out_y, xz, gamma, beta, yg);
    // 6. out_proj: out = yg @ out_proj_w^T  (8192 x 96, K=192)
    {
        int waves = (M / 16) * (96 / 16);      // 3072
        gemm_wmma_xt<<<waves / 8, 256, 0, stream>>>(yg, out_proj_w, out,
                                                    M, DMODEL, DINNER, DINNER, DINNER, DMODEL);
    }
}
